// NGP_40097814675941
// MI455X (gfx1250) — compile-verified, matrix operand order checked by reading, not run on hardware
//
#include <hip/hip_runtime.h>
#include <stdint.h>

typedef __attribute__((ext_vector_type(16))) _Float16 v16h;
typedef __attribute__((ext_vector_type(8)))  float    v8f;

union FragU { uint32_t u[8]; v16h h; };
union H2U   { _Float16 h[2]; uint32_t u; };

#define LOG2T 19
#define TSIZE (1u << LOG2T)   // hash table entries per level

__device__ __forceinline__ v8f wmma_f16(v16h a, v16h b, v8f c) {
  // D(16x16 f32) = A(16x32 f16) * B(32x16 f16) + C
  return __builtin_amdgcn_wmma_f32_16x16x32_f16(false, a, false, b, (short)0, c,
                                                false, false);
}

// K index held by (lane-half lh, vgpr v, part p) for a 16-bit 16x32 A operand
// (ISA 7.12.2): lanes 0-15: V0-3 -> K 0..7, V4-7 -> K 16..23 ; lanes 16-31: +8.
__device__ __forceinline__ int kmap(int v, int lh) {
  return ((v < 4) ? (2 * v) : (2 * v + 8)) + lh * 8;
}

__device__ __forceinline__ v16h load_wfrag(const uint32_t* p) {
  FragU fu;
  const uint4* q = (const uint4*)p;
  uint4 a = q[0], b = q[1];
  fu.u[0] = a.x; fu.u[1] = a.y; fu.u[2] = a.z; fu.u[3] = a.w;
  fu.u[4] = b.x; fu.u[5] = b.y; fu.u[6] = b.z; fu.u[7] = b.w;
  return fu.h;
}

// Assemble an A fragment (rows = points, K = feature dim) from an LDS f16 tile.
__device__ __forceinline__ v16h load_act_frag(const _Float16* buf, int stride,
                                              int lane, int kbase) {
  int mm = lane & 15, lh = lane >> 4;
  FragU fu;
#pragma unroll
  for (int v = 0; v < 8; ++v) {
    int k = kbase + kmap(v, lh);
    fu.u[v] = *(const uint32_t*)(buf + mm * stride + k);  // 2 consecutive halves
  }
  return fu.h;
}

// Scatter a 16x16 f32 D tile into an LDS f16 activation tile at column offset.
__device__ __forceinline__ void store_tile(_Float16* buf, int stride, int lane,
                                           int coloff, v8f d) {
  int nn = lane & 15;
  int rb = (lane >> 4) * 8;   // lanes 0-15: rows 0-7, lanes 16-31: rows 8-15
#pragma unroll
  for (int r = 0; r < 8; ++r)
    buf[(rb + r) * stride + coloff + nn] = (_Float16)d[r];
}

__global__ __launch_bounds__(256) void ngp_fused(
    const float* __restrict__ x, const float* __restrict__ dirs,
    const float* __restrict__ hash_table,
    const float* __restrict__ w1, const float* __restrict__ w2,
    const float* __restrict__ w3, const float* __restrict__ w4,
    const float* __restrict__ w5,
    float* __restrict__ out, int npts, int tiles) {
  // ---- LDS: pre-swizzled f16 weight fragments + per-wave activation tiles
  __shared__ __align__(16) uint32_t wfrag[20][32][8];     // 20 KB
  __shared__ __align__(16) _Float16 actR[8][16][32];      // enc / rgb_in (8 KB)
  __shared__ __align__(16) _Float16 actH[8][16][64];      // h1 / a / a2 (16 KB)

  // ---- once per block: convert f32 weights -> f16 B fragments ("B loaded as
  //      A of B^T": fragment element (lane,h) = W[koff + kmap][noff + (lane&15)])
  for (int e = threadIdx.x; e < 20 * 32; e += blockDim.x) {
    int f = e >> 5, lane = e & 31;
    const float* W; int Nd, koff, noff, nvalid;
    if (f < 4)       { W = w1; Nd = 64; koff = 0;              noff = 16 * f;            nvalid = 16; }
    else if (f < 6)  { W = w2; Nd = 16; koff = 32 * (f - 4);   noff = 0;                 nvalid = 16; }
    else if (f < 10) { W = w3; Nd = 64; koff = 0;              noff = 16 * (f - 6);      nvalid = 16; }
    else if (f < 18) { W = w4; Nd = 64; koff = 32*((f-10)>>2); noff = 16 * ((f-10) & 3); nvalid = 16; }
    else             { W = w5; Nd = 3;  koff = 32 * (f - 18);  noff = 0;                 nvalid = 3;  }
    int mm = lane & 15, lh = lane >> 4;
#pragma unroll
    for (int v = 0; v < 8; ++v) {
      H2U pk;
#pragma unroll
      for (int p = 0; p < 2; ++p) {
        int k = koff + kmap(v, lh) + p;
        float val = (mm < nvalid) ? W[k * Nd + noff + mm] : 0.0f;  // pad w5 cols
        pk.h[p] = (_Float16)val;
      }
      wfrag[f][lane][v] = pk.u;
    }
  }
  __syncthreads();

  const int lane = threadIdx.x & 31;
  const int wv   = threadIdx.x >> 5;
  const int m    = lane & 15;    // point within tile
  const int s    = lane >> 4;    // which half of levels / SH coeffs
  _Float16* rgbBuf = &actR[wv][0][0];   // 16x32: enc, later sh|h
  _Float16* hBuf   = &actH[wv][0][0];   // 16x64: h1, later a, a2
  const float2* __restrict__ ht2 = (const float2*)hash_table;  // uniform base

  static const int RES_LO[8] = {16, 21, 27, 36, 48, 64, 84, 111};
  static const int RES_HI[8] = {147, 194, 256, 337, 445, 588, 776, 1024};

  const int gw = blockIdx.x * 8 + wv;
  const int nw = gridDim.x * 8;
  for (int tile = gw; tile < tiles; tile += nw) {
    int pt = tile * 16 + m;
    // ================= hash encode (lane does 8 levels of its point) ========
    float x0 = x[3 * pt + 0] + 0.5f;   // (x+SCALE)/(2*SCALE) with SCALE=0.5
    float y0 = x[3 * pt + 1] + 0.5f;
    float z0 = x[3 * pt + 2] + 0.5f;
#pragma unroll
    for (int lv = 0; lv < 8; ++lv) {
      float res = (float)(s ? RES_HI[lv] : RES_LO[lv]);
      uint32_t lvlbase = ((uint32_t)(s * 8 + lv)) << LOG2T;  // level * TSIZE
      float px = x0 * res, py = y0 * res, pz = z0 * res;
      float fx = floorf(px), fy = floorf(py), fz = floorf(pz);
      uint32_t cx = (uint32_t)fx, cy = (uint32_t)fy, cz = (uint32_t)fz;
      float wx1 = px - fx, wy1 = py - fy, wz1 = pz - fz;
      float wx0 = 1.0f - wx1, wy0 = 1.0f - wy1, wz0 = 1.0f - wz1;
      // strength-reduced hash terms (1 xor+xor+and+add per corner)
      uint32_t hy0 = cy * 2654435761u, hy1 = hy0 + 2654435761u;
      uint32_t hz0 = cz * 805459861u,  hz1 = hz0 + 805459861u;
      float wyz00 = wy0 * wz0, wyz01 = wy0 * wz1;
      float wyz10 = wy1 * wz0, wyz11 = wy1 * wz1;
      float f0 = 0.0f, f1 = 0.0f;
#pragma unroll
      for (int c = 0; c < 8; ++c) {
        uint32_t ci = (c >> 2) & 1, cj = (c >> 1) & 1, ck = c & 1;
        uint32_t h = (cx + ci) ^ (cj ? hy1 : hy0) ^ (ck ? hz1 : hz0);
        uint32_t eidx = lvlbase + (h & (TSIZE - 1u));   // 32-bit element index
        float2 f = ht2[eidx];                           // saddr + voffset gather
        float w = (ci ? wx1 : wx0) *
                  (cj ? (ck ? wyz11 : wyz10) : (ck ? wyz01 : wyz00));
        f0 += w * f.x; f1 += w * f.y;
      }
      H2U pk; pk.h[0] = (_Float16)f0; pk.h[1] = (_Float16)f1;
      *(uint32_t*)&rgbBuf[m * 32 + 2 * (s * 8 + lv)] = pk.u;
    }
    // ================= density MLP ==========================================
    v16h encA = load_act_frag(rgbBuf, 32, lane, 0);
#pragma unroll
    for (int nc = 0; nc < 4; ++nc) {          // enc(16x32) @ w1 -> relu -> h1
      v8f acc = {};
      acc = wmma_f16(encA, load_wfrag(&wfrag[nc][lane][0]), acc);
#pragma unroll
      for (int r = 0; r < 8; ++r) acc[r] = fmaxf(acc[r], 0.0f);
      store_tile(hBuf, 64, lane, nc * 16, acc);
    }
    v16h a0 = load_act_frag(hBuf, 64, lane, 0);
    v16h a1 = load_act_frag(hBuf, 64, lane, 32);
    v8f hAcc = {};                            // h = h1 @ w2 (no relu)
    hAcc = wmma_f16(a0, load_wfrag(&wfrag[4][lane][0]), hAcc);
    hAcc = wmma_f16(a1, load_wfrag(&wfrag[5][lane][0]), hAcc);
    if (m == 0) {                             // sigma = exp(h[:,0]) (col 0)
#pragma unroll
      for (int r = 0; r < 8; ++r)
        out[tile * 16 + s * 8 + r] = __expf(hAcc[r]);
    }
    store_tile(rgbBuf, 32, lane, 16, hAcc);   // h -> rgb_in cols 16..31
    // ================= SH deg 4 -> rgb_in cols 0..15 ========================
    float ddx = dirs[3 * pt + 0], ddy = dirs[3 * pt + 1], ddz = dirs[3 * pt + 2];
    float rinv = rsqrtf(ddx * ddx + ddy * ddy + ddz * ddz);
    float sx = ddx * rinv, sy = ddy * rinv, sz = ddz * rinv;
    float xy = sx * sy, xz = sx * sz, yz = sy * sz;
    float x2 = sx * sx, y2 = sy * sy, z2 = sz * sz;
    float c0 = 0.28209479177387814f;
    float c1 = -0.48860251190291987f * sy;
    float c2 =  0.48860251190291987f * sz;
    float c3 = -0.48860251190291987f * sx;
    float c4 =  1.0925484305920792f * xy;
    float c5 = -1.0925484305920792f * yz;
    float c6 =  0.94617469575756f * z2 - 0.31539156525252f;
    float c7 = -1.0925484305920792f * xz;
    float c8 =  0.5462742152960396f * (x2 - y2);
    float c9 =  0.5900435899266435f * sy * (-3.0f * x2 + y2);
    float c10 = 2.890611442640554f * xy * sz;
    float c11 = 0.4570457994644657f * sy * (1.0f - 5.0f * z2);
    float c12 = 0.3731763325901154f * sz * (5.0f * z2 - 3.0f);
    float c13 = 0.4570457994644657f * sx * (1.0f - 5.0f * z2);
    float c14 = 1.445305721320277f * sz * (x2 - y2);
    float c15 = 0.5900435899266435f * sx * (-x2 + 3.0f * y2);
    float o0 = s ? c8  : c0, o1 = s ? c9  : c1, o2 = s ? c10 : c2, o3 = s ? c11 : c3;
    float o4 = s ? c12 : c4, o5 = s ? c13 : c5, o6 = s ? c14 : c6, o7 = s ? c15 : c7;
    {
      H2U p0; p0.h[0] = (_Float16)o0; p0.h[1] = (_Float16)o1;
      H2U p1; p1.h[0] = (_Float16)o2; p1.h[1] = (_Float16)o3;
      H2U p2; p2.h[0] = (_Float16)o4; p2.h[1] = (_Float16)o5;
      H2U p3; p3.h[0] = (_Float16)o6; p3.h[1] = (_Float16)o7;
      uint32_t* dst = (uint32_t*)&rgbBuf[m * 32 + s * 8];
      dst[0] = p0.u; dst[1] = p1.u; dst[2] = p2.u; dst[3] = p3.u;
    }
    // ================= color MLP ============================================
    v16h rgbA = load_act_frag(rgbBuf, 32, lane, 0);
#pragma unroll
    for (int nc = 0; nc < 4; ++nc) {          // rgb_in(16x32) @ w3 -> relu -> a
      v8f acc = {};
      acc = wmma_f16(rgbA, load_wfrag(&wfrag[6 + nc][lane][0]), acc);
#pragma unroll
      for (int r = 0; r < 8; ++r) acc[r] = fmaxf(acc[r], 0.0f);
      store_tile(hBuf, 64, lane, nc * 16, acc);
    }
    a0 = load_act_frag(hBuf, 64, lane, 0);
    a1 = load_act_frag(hBuf, 64, lane, 32);
#pragma unroll
    for (int nc = 0; nc < 4; ++nc) {          // a(16x64) @ w4 -> relu -> a2
      v8f acc = {};
      acc = wmma_f16(a0, load_wfrag(&wfrag[10 + nc][lane][0]), acc);
      acc = wmma_f16(a1, load_wfrag(&wfrag[14 + nc][lane][0]), acc);
#pragma unroll
      for (int r = 0; r < 8; ++r) acc[r] = fmaxf(acc[r], 0.0f);
      store_tile(hBuf, 64, lane, nc * 16, acc);
    }
    a0 = load_act_frag(hBuf, 64, lane, 0);
    a1 = load_act_frag(hBuf, 64, lane, 32);
    v8f oAcc = {};                            // a2(16x64) @ w5 (padded 64x16)
    oAcc = wmma_f16(a0, load_wfrag(&wfrag[18][lane][0]), oAcc);
    oAcc = wmma_f16(a1, load_wfrag(&wfrag[19][lane][0]), oAcc);
    if (m < 3) {                              // cols 0..2 valid -> sigmoid
#pragma unroll
      for (int r = 0; r < 8; ++r) {
        float v = oAcc[r];
        out[npts + 3 * (tile * 16 + s * 8 + r) + m] = 1.0f / (1.0f + __expf(-v));
      }
    }
  }
}

extern "C" void kernel_launch(void* const* d_in, const int* in_sizes, int n_in,
                              void* d_out, int out_size, void* d_ws, size_t ws_size,
                              hipStream_t stream) {
  (void)n_in; (void)out_size; (void)d_ws; (void)ws_size;
  const float* x  = (const float*)d_in[0];
  const float* d  = (const float*)d_in[1];
  const float* ht = (const float*)d_in[2];
  const float* w1 = (const float*)d_in[3];
  const float* w2 = (const float*)d_in[4];
  const float* w3 = (const float*)d_in[5];
  const float* w4 = (const float*)d_in[6];
  const float* w5 = (const float*)d_in[7];
  float* out = (float*)d_out;
  int npts  = in_sizes[0] / 3;
  int tiles = npts / 16;
  int blocks = (tiles + 7) / 8;
  if (blocks > 2048) blocks = 2048;
  if (blocks < 1) blocks = 1;
  ngp_fused<<<blocks, 256, 0, stream>>>(x, d, ht, w1, w2, w3, w4, w5,
                                        out, npts, tiles);
}
